// Transformer_15169824489573
// MI455X (gfx1250) — compile-verified, matrix operand order checked by reading, not run on hardware
//
#include <hip/hip_runtime.h>
#include <hip/hip_bf16.h>

// Transformer forward, MI455X (gfx1250). All large GEMMs via
// v_wmma_f32_16x16x32_bf16 (wave32, fp32 accum). fp32 weights/activations are
// converted to bf16 in-register while staging tiles into LDS.
// Staging uses per-thread incremental pointers (no per-step 64-bit muls) and
// global_prefetch two k-steps ahead of the double-buffered loads.

#define NB      4
#define TSEQ    512
#define DMODEL  1024
#define NHEAD   16
#define NLAYER  8
#define VOCAB   32000
#define HDIM    64
#define DFF     4096

typedef __attribute__((ext_vector_type(16))) __bf16       v16bf;
typedef __attribute__((ext_vector_type(8)))  float        v8f;
typedef __attribute__((ext_vector_type(4)))  unsigned int u32x4;

union Frag { v16bf v; u32x4 u[2]; };

__device__ __forceinline__ unsigned short f2bf(float a) {
  union { float f; unsigned int u; } x; x.f = a;
  unsigned int u = x.u + (0x7fffu + ((x.u >> 16) & 1u));   // round-nearest-even
  return (unsigned short)(u >> 16);
}
__device__ __forceinline__ unsigned int f2bf2(float a, float b) {
  union { float f; unsigned int u; } x, y; x.f = a; y.f = b;
  unsigned int ua = x.u + (0x7fffu + ((x.u >> 16) & 1u));
  unsigned int ub = y.u + (0x7fffu + ((y.u >> 16) & 1u));
  return (ua >> 16) | (ub & 0xffff0000u);
}

// ---------------------------------------------------------------------------
// Generic tiled WMMA GEMM:  C = act(alpha * A(MxK) * B(KxN) + bias)
//   A fp32 row-major (lda), B fp32:  TRANSB=false -> B[k*ldb + n]
//                                    TRANSB=true  -> B[n*ldb + k]
//   Optional batching over grid.z: z -> (n = z/bH, h = z%bH) pointer offsets.
//   BMxBN tile, BK=32 (one bf16 WMMA K-step), 256 threads = 8 waves,
//   each wave computes WMxWN via (WM/16)x(WN/16) 16x16x32 WMMA tiles.
//   Double-buffered LDS, bf16 tiles, B stored K-innermost for b128 frag loads.
// ---------------------------------------------------------------------------
template <int BM, int BN, int WM, int WN, bool TRANSB>
__global__ __launch_bounds__(256) void wmma_gemm(
    const float* __restrict__ A, const float* __restrict__ B,
    const float* __restrict__ bias, float* __restrict__ C,
    int K, int lda, int ldb, int ldc,
    int bH, long long sAn, long long sAh, long long sBn, long long sBh,
    long long sCn, long long sCh, float alpha, int relu)
{
  constexpr int BK = 32;
  constexpr int LK = BK + 8;                       // pad: bank-conflict relief
  __shared__ __attribute__((aligned(16))) unsigned short As[2][BM][LK];
  __shared__ __attribute__((aligned(16))) unsigned short Bs[2][BN][LK];

  if (bH > 0) {                                    // batched (attention) mode
    const int z = blockIdx.z;
    const int n = z / bH, h = z - n * bH;
    A += n * sAn + h * sAh;
    B += n * sBn + h * sBh;
    C += n * sCn + h * sCh;
  }
  const long long tileM = (long long)blockIdx.y * BM;
  const long long tileN = (long long)blockIdx.x * BN;
  A += tileM * lda;
  if (TRANSB) B += tileN * ldb; else B += tileN;

  const int tid  = threadIdx.x;
  const int wid  = tid >> 5, lane = tid & 31;
  constexpr int WGN = BN / WN;
  const int wr = wid / WGN, wc = wid - wr * WGN;
  constexpr int MT  = WM / 16, NTL = WN / 16;
  const int lmod = lane & 15;
  const int kb   = (lane >> 4) * 8;                // lane-half K base (ISA layout)

  // ---- per-thread incremental staging pointers (hoisted address math) ----
  constexpr int A_IT = (BM * BK) / (4 * 256);
  constexpr int B_IT = (BN * BK) / (4 * 256);      // same count in both modes
  const float* aSrc[A_IT];
  int aR[A_IT], aC[A_IT];
  #pragma unroll
  for (int i = 0; i < A_IT; ++i) {
    const int f = tid + i * 256;
    aR[i] = f >> 3;                                // BK/4 == 8 float4 per row
    aC[i] = (f & 7) << 2;
    aSrc[i] = A + (long long)aR[i] * lda + aC[i];
  }
  const float* bSrc[B_IT];
  int bR[B_IT], bC[B_IT];
  const long long bStep = TRANSB ? (long long)BK : (long long)BK * ldb;
  #pragma unroll
  for (int i = 0; i < B_IT; ++i) {
    const int f = tid + i * 256;
    if (TRANSB) {
      bR[i] = f >> 3;                              // n row
      bC[i] = (f & 7) << 2;                        // k col
      bSrc[i] = B + (long long)bR[i] * ldb + bC[i];
    } else {
      bR[i] = f / (BN / 4);                        // k row
      bC[i] = (f - bR[i] * (BN / 4)) << 2;         // n col
      bSrc[i] = B + (long long)bR[i] * ldb + bC[i];
    }
  }

  v8f acc[MT][NTL];
  #pragma unroll
  for (int mi = 0; mi < MT; ++mi)
    #pragma unroll
    for (int ni = 0; ni < NTL; ++ni)
      #pragma unroll
      for (int e = 0; e < 8; ++e) acc[mi][ni][e] = 0.0f;

  auto stage = [&](int buf) {
    #pragma unroll
    for (int i = 0; i < A_IT; ++i) {
      const float* s = aSrc[i];
      aSrc[i] = s + BK;
      __builtin_prefetch(s + 2 * BK, 0, 0);        // k-step + 2 -> L2
      unsigned int* d = (unsigned int*)&As[buf][aR[i]][aC[i]];
      const float x0 = s[0], x1 = s[1], x2 = s[2], x3 = s[3];
      d[0] = f2bf2(x0, x1);
      d[1] = f2bf2(x2, x3);
    }
    #pragma unroll
    for (int i = 0; i < B_IT; ++i) {
      const float* s = bSrc[i];
      bSrc[i] = s + bStep;
      __builtin_prefetch(s + 2 * bStep, 0, 0);
      const float x0 = s[0], x1 = s[1], x2 = s[2], x3 = s[3];
      if (TRANSB) {                                // direct: rows are N, cols K
        unsigned int* d = (unsigned int*)&Bs[buf][bR[i]][bC[i]];
        d[0] = f2bf2(x0, x1);
        d[1] = f2bf2(x2, x3);
      } else {                                     // transpose into LDS
        Bs[buf][bC[i] + 0][bR[i]] = f2bf(x0);
        Bs[buf][bC[i] + 1][bR[i]] = f2bf(x1);
        Bs[buf][bC[i] + 2][bR[i]] = f2bf(x2);
        Bs[buf][bC[i] + 3][bR[i]] = f2bf(x3);
      }
    }
  };

  auto compute = [&](int buf) {
    Frag a[MT], b[NTL];
    #pragma unroll
    for (int mi = 0; mi < MT; ++mi) {
      const unsigned short* p = &As[buf][wr * WM + mi * 16 + lmod][kb];
      a[mi].u[0] = *(const u32x4*)(p);             // K kb..kb+7
      a[mi].u[1] = *(const u32x4*)(p + 16);        // K kb+16..kb+23
    }
    #pragma unroll
    for (int ni = 0; ni < NTL; ++ni) {
      const unsigned short* p = &Bs[buf][wc * WN + ni * 16 + lmod][kb];
      b[ni].u[0] = *(const u32x4*)(p);
      b[ni].u[1] = *(const u32x4*)(p + 16);
    }
    #pragma unroll
    for (int mi = 0; mi < MT; ++mi)
      #pragma unroll
      for (int ni = 0; ni < NTL; ++ni)
        acc[mi][ni] = __builtin_amdgcn_wmma_f32_16x16x32_bf16(
            false, a[mi].v, false, b[ni].v, (short)0, acc[mi][ni], false, false);
  };

  const int nk = K / BK;
  stage(0);
  __syncthreads();
  for (int kt = 0; kt < nk; ++kt) {
    const int cur = kt & 1;
    if (kt + 1 < nk) stage(cur ^ 1);
    compute(cur);
    __syncthreads();
  }

  // Epilogue: C/D 16x16 layout -> VGPR e: M = e + 8*(lane>=16), N = lane&15
  #pragma unroll
  for (int mi = 0; mi < MT; ++mi) {
    const int m0 = wr * WM + mi * 16 + (lane >> 4) * 8;
    #pragma unroll
    for (int ni = 0; ni < NTL; ++ni) {
      const long long gn = tileN + wc * WN + ni * 16 + lmod;
      const float bv = bias ? bias[gn] : 0.0f;
      float* cp = C + (tileM + m0) * (long long)ldc + gn;
      #pragma unroll
      for (int e = 0; e < 8; ++e) {
        float vv = alpha * acc[mi][ni][e] + bv;
        if (relu) vv = fmaxf(vv, 0.0f);
        cp[(long long)e * ldc] = vv;
      }
    }
  }
}

// ---------------------------------------------------------------------------
// x[n,t,:] = emb[tokens[n,t],:] + pe[n,:]   (reference indexes pe by batch!)
// ---------------------------------------------------------------------------
__global__ __launch_bounds__(256) void embed_kernel(
    const int* __restrict__ tokens, const float* __restrict__ emb,
    const float* __restrict__ pe, float* __restrict__ x)
{
  const int nt  = blockIdx.x;
  const int n   = nt / TSEQ;
  const int tok = tokens[nt];
  const float* e = emb + (size_t)tok * DMODEL;
  const float* p = pe  + (size_t)n   * DMODEL;
  float* o = x + (size_t)nt * DMODEL;
  for (int d = threadIdx.x; d < DMODEL; d += 256) o[d] = e[d] + p[d];
}

// ---------------------------------------------------------------------------
// Causal softmax over rows of S[(n*H+h)*T + t][0..T). One wave per row.
// ---------------------------------------------------------------------------
__global__ __launch_bounds__(256) void softmax_causal_kernel(float* __restrict__ S)
{
  const int wid = threadIdx.x >> 5, lane = threadIdx.x & 31;
  const long long row = (long long)blockIdx.x * 8 + wid;
  const int t = (int)(row & (TSEQ - 1));
  float* p = S + row * (long long)TSEQ;
  float v[TSEQ / 32];
  float mx = -3.0e38f;
  #pragma unroll
  for (int i = 0; i < TSEQ / 32; ++i) {
    const int c = lane + i * 32;
    const float x = p[c];
    v[i] = (c <= t) ? x : -3.0e38f;
    mx = fmaxf(mx, v[i]);
  }
  #pragma unroll
  for (int off = 16; off; off >>= 1) mx = fmaxf(mx, __shfl_xor(mx, off, 32));
  float sum = 0.0f;
  #pragma unroll
  for (int i = 0; i < TSEQ / 32; ++i) {
    const int c = lane + i * 32;
    v[i] = (c <= t) ? __expf(v[i] - mx) : 0.0f;
    sum += v[i];
  }
  #pragma unroll
  for (int off = 16; off; off >>= 1) sum += __shfl_xor(sum, off, 32);
  const float inv = 1.0f / sum;
  #pragma unroll
  for (int i = 0; i < TSEQ / 32; ++i) p[lane + i * 32] = v[i] * inv;
}

// ---------------------------------------------------------------------------
// h[row,:] += LayerNorm(f[row,:]) * g + b    (one 256-thread block per row)
// ---------------------------------------------------------------------------
__global__ __launch_bounds__(256) void ln_residual_kernel(
    float* __restrict__ h, const float* __restrict__ f,
    const float* __restrict__ g, const float* __restrict__ b)
{
  const int row = blockIdx.x;
  const float* fr = f + (size_t)row * DMODEL;
  float* hr = h + (size_t)row * DMODEL;
  float v[DMODEL / 256];
  float s = 0.0f, ss = 0.0f;
  #pragma unroll
  for (int i = 0; i < DMODEL / 256; ++i) {
    v[i] = fr[threadIdx.x + i * 256];
    s += v[i]; ss += v[i] * v[i];
  }
  #pragma unroll
  for (int off = 16; off; off >>= 1) {
    s += __shfl_xor(s, off, 32); ss += __shfl_xor(ss, off, 32);
  }
  __shared__ float sh1[8], sh2[8];
  const int wid = threadIdx.x >> 5, lane = threadIdx.x & 31;
  if (lane == 0) { sh1[wid] = s; sh2[wid] = ss; }
  __syncthreads();
  if (wid == 0) {
    s  = (lane < 8) ? sh1[lane] : 0.0f;
    ss = (lane < 8) ? sh2[lane] : 0.0f;
    #pragma unroll
    for (int off = 4; off; off >>= 1) {
      s += __shfl_xor(s, off, 32); ss += __shfl_xor(ss, off, 32);
    }
    if (lane == 0) {
      const float m = s / DMODEL;
      sh1[0] = m;
      sh2[0] = rsqrtf(ss / DMODEL - m * m + 1e-5f);
    }
  }
  __syncthreads();
  const float m = sh1[0], inv = sh2[0];
  #pragma unroll
  for (int i = 0; i < DMODEL / 256; ++i) {
    const int c = threadIdx.x + i * 256;
    hr[c] += (v[i] - m) * inv * g[c] + b[c];
  }
}

// ---------------------------------------------------------------------------
extern "C" void kernel_launch(void* const* d_in, const int* in_sizes, int n_in,
                              void* d_out, int out_size, void* d_ws, size_t ws_size,
                              hipStream_t stream)
{
  (void)in_sizes; (void)n_in; (void)out_size;
  const int*   tokens = (const int*)  d_in[0];
  const float* emb    = (const float*)d_in[1];
  const float* pe     = (const float*)d_in[2];
  const float* Wq     = (const float*)d_in[3];
  const float* bq     = (const float*)d_in[4];
  const float* Wk     = (const float*)d_in[5];
  const float* bk     = (const float*)d_in[6];
  const float* Wv     = (const float*)d_in[7];
  const float* bv     = (const float*)d_in[8];
  const float* Wo     = (const float*)d_in[9];
  const float* bo     = (const float*)d_in[10];
  const float* g1     = (const float*)d_in[11];
  const float* be1    = (const float*)d_in[12];
  const float* W1     = (const float*)d_in[13];
  const float* b1     = (const float*)d_in[14];
  const float* W2     = (const float*)d_in[15];
  const float* b2     = (const float*)d_in[16];
  const float* g2     = (const float*)d_in[17];
  const float* be2    = (const float*)d_in[18];
  const float* Wfc    = (const float*)d_in[19];
  const float* bfc    = (const float*)d_in[20];

  const size_t NTD  = (size_t)NB * TSEQ * DMODEL;        // 2 M
  const size_t NHTT = (size_t)NB * NHEAD * TSEQ * TSEQ;  // 16.78 M
  const size_t NMID = (size_t)NB * TSEQ * DFF;           // 8 M
  if (ws_size < (5 * NTD + NHTT + NMID) * sizeof(float)) return;

  float* x    = (float*)d_ws;      // hidden state (residual stream)
  float* qb   = x    + NTD;        // Q / generic scratch
  float* kbuf = qb   + NTD;        // K
  float* vb   = kbuf + NTD;        // V
  float* ob   = vb   + NTD;        // attention output
  float* Sb   = ob   + NTD;        // attention scores/probs (N*H, T, T)
  float* mid  = Sb   + NHTT;       // FFN hidden

  const int NT = NB * TSEQ;        // 2048 token rows
  const long long sTD = (long long)TSEQ * DMODEL;
  const long long sTT = (long long)TSEQ * TSEQ;

  embed_kernel<<<NT, 256, 0, stream>>>(tokens, emb, pe, x);

  auto gemm_flat = [&](const float* A, const float* B, const float* bias, float* C,
                       int M, int Nc, int K, int relu) {
    dim3 g(Nc / 128, M / 128, 1);
    wmma_gemm<128, 128, 64, 32, false><<<g, 256, 0, stream>>>(
        A, B, bias, C, K, K, Nc, Nc, 0, 0, 0, 0, 0, 0, 0, 1.0f, relu);
  };

  for (int l = 0; l < NLAYER; ++l) {
    const float* wq  = Wq  + (size_t)l * DMODEL * DMODEL;
    const float* wk  = Wk  + (size_t)l * DMODEL * DMODEL;
    const float* wv  = Wv  + (size_t)l * DMODEL * DMODEL;
    const float* wo  = Wo  + (size_t)l * DMODEL * DMODEL;
    const float* w1  = W1  + (size_t)l * DMODEL * DFF;
    const float* w2  = W2  + (size_t)l * DFF * DMODEL;
    const float* bq_ = bq  + (size_t)l * DMODEL;
    const float* bk_ = bk  + (size_t)l * DMODEL;
    const float* bv_ = bv  + (size_t)l * DMODEL;
    const float* bo_ = bo  + (size_t)l * DMODEL;
    const float* b1_ = b1  + (size_t)l * DFF;
    const float* b2_ = b2  + (size_t)l * DMODEL;
    const float* g1_ = g1  + (size_t)l * DMODEL;
    const float* e1_ = be1 + (size_t)l * DMODEL;
    const float* g2_ = g2  + (size_t)l * DMODEL;
    const float* e2_ = be2 + (size_t)l * DMODEL;

    // QKV projections over all 2048 token rows at once
    gemm_flat(x, wq, bq_, qb,   NT, DMODEL, DMODEL, 0);
    gemm_flat(x, wk, bk_, kbuf, NT, DMODEL, DMODEL, 0);
    gemm_flat(x, wv, bv_, vb,   NT, DMODEL, DMODEL, 0);

    // S = (1/sqrt(HD)) * Q K^T, batched over (n,h); B is K-matrix, N-major
    {
      dim3 g(TSEQ / 128, TSEQ / 128, NB * NHEAD);
      wmma_gemm<128, 128, 64, 32, true><<<g, 256, 0, stream>>>(
          qb, kbuf, nullptr, Sb, HDIM, DMODEL, DMODEL, TSEQ,
          NHEAD, sTD, HDIM, sTD, HDIM,
          (long long)NHEAD * sTT, sTT, 0.125f, 0);
    }

    softmax_causal_kernel<<<(NB * NHEAD * TSEQ) / 8, 256, 0, stream>>>(Sb);

    // O = P V, batched (M=512, N=64, K=512) -> 64x64 tile config
    {
      dim3 g(HDIM / 64, TSEQ / 64, NB * NHEAD);
      wmma_gemm<64, 64, 32, 16, false><<<g, 256, 0, stream>>>(
          Sb, vb, nullptr, ob, TSEQ, TSEQ, DMODEL, DMODEL,
          NHEAD, (long long)NHEAD * sTT, sTT, sTD, HDIM, sTD, HDIM, 1.0f, 0);
    }

    // Output projection + residual LayerNorm
    gemm_flat(ob, wo, bo_, qb, NT, DMODEL, DMODEL, 0);
    ln_residual_kernel<<<NT, 256, 0, stream>>>(x, qb, g1_, e1_);

    // FFN: relu(x W1 + b1) W2 + b2, then residual LayerNorm
    gemm_flat(x,   w1, b1_, mid, NT, DFF,    DMODEL, 1);
    gemm_flat(mid, w2, b2_, qb,  NT, DMODEL, DFF,    0);
    ln_residual_kernel<<<NT, 256, 0, stream>>>(x, qb, g2_, e2_);
  }

  // Logits: (2048 x 1024) @ (1024 x 32000) + bfc -> d_out
  gemm_flat(x, Wfc, bfc, (float*)d_out, NT, VOCAB, DMODEL, 0);
}
